// LanguageModel_33775622815826
// MI455X (gfx1250) — matrix operand order repaired and checked by reading, not compiled, not run on hardware
//
#include <hip/hip_runtime.h>

// ---------------------------------------------------------------------------
// GPT forward for MI455X (gfx1250, wave32, WMMA).
// All GEMMs run on v_wmma_f32_16x16x32_bf16 with double-buffered fp32->bf16
// LDS staging (global loads overlap the matrix pipe; one barrier per K-step).
// ---------------------------------------------------------------------------

typedef __bf16 bf16;
typedef __attribute__((ext_vector_type(16))) __bf16 v16bf;
typedef __attribute__((ext_vector_type(8)))  float  v8f;

#define E_DIM   1024
#define FF_DIM  4096
#define V_DIM   50257
#define L_NUM   12
#define T_SEQ   128
#define B_SZ    16
#define M_ROWS  2048          // B*T
#define H_NUM   16
#define D_HEAD  64

// ---------------------------------------------------------------------------
// WMMA fragment loaders (layouts per CDNA5 ISA 7.12.2, wave32).
//  A (16x32 bf16): lane 0..15 -> M=lane, K half kh=8*(lane>>4);
//    regs 0..7 hold K = kh..kh+7 and kh+16..kh+23 (two 16B contiguous chunks).
//  B (32x16 bf16): lane 0..15 -> N=lane, same K order as A (kh..kh+7, kh+16..kh+23).
//  C/D (16x16 f32): reg r, lanes 0..15 -> M=r,N=lane; lanes 16..31 -> M=r+8.
// ---------------------------------------------------------------------------
union FragU { v16bf v; uint4 q[2]; };

__device__ __forceinline__ v16bf load_frag_a(const bf16* a, int lda) {
  const int lane = threadIdx.x & 31;
  const int m    = lane & 15;
  const int kh   = (lane >> 4) * 8;
  const bf16* p  = a + (size_t)m * lda + kh;
  FragU u;
  u.q[0] = *(const uint4*)(p);
  u.q[1] = *(const uint4*)(p + 16);
  return u.v;
}

__device__ __forceinline__ v16bf load_frag_b(const bf16* b, int ldb, int nb) {
  const int lane = threadIdx.x & 31;
  const int kh   = (lane >> 4) * 8;
  const bf16* p  = b + (size_t)kh * ldb + nb + (lane & 15);
  v16bf v;
  #pragma unroll
  for (int i = 0; i < 8; ++i) {
    v[i]     = p[(size_t)i * ldb];
    v[8 + i] = p[(size_t)(16 + i) * ldb];
  }
  return v;
}

__device__ __forceinline__ v8f wmma_bf16(v16bf a, v16bf b, v8f c) {
  return __builtin_amdgcn_wmma_f32_16x16x32_bf16(false, a, false, b,
                                                 (short)0, c, false, false);
}

// ---------------------------------------------------------------------------
// Embedding gather: x[bt] = tok_emb[idx[bt]] + pos_emb[t]
// ---------------------------------------------------------------------------
__global__ __launch_bounds__(256)
void embed_kernel(const int* __restrict__ idx, const float* __restrict__ tok,
                  const float* __restrict__ pos, float* __restrict__ x) {
  const int row = blockIdx.x;          // b*T + t
  const int t   = row & (T_SEQ - 1);
  const int id  = idx[row];
  const float* te = tok + (size_t)id * E_DIM;
  const float* pe = pos + (size_t)t * E_DIM;
  float* xr = x + (size_t)row * E_DIM;
  for (int i = threadIdx.x; i < E_DIM; i += 256) xr[i] = te[i] + pe[i];
}

// ---------------------------------------------------------------------------
// Row LayerNorm over E=1024, one block per row.
// ---------------------------------------------------------------------------
__global__ __launch_bounds__(256)
void layernorm_kernel(const float* __restrict__ x, const float* __restrict__ g,
                      const float* __restrict__ b, float* __restrict__ y) {
  __shared__ float red[16];
  const int row = blockIdx.x;
  const float* xr = x + (size_t)row * E_DIM;
  float s = 0.f, s2 = 0.f;
  for (int i = threadIdx.x; i < E_DIM; i += 256) {
    float v = xr[i]; s += v; s2 += v * v;
  }
  #pragma unroll
  for (int m = 16; m >= 1; m >>= 1) {
    s  += __shfl_xor(s,  m, 32);
    s2 += __shfl_xor(s2, m, 32);
  }
  const int w = threadIdx.x >> 5;
  if ((threadIdx.x & 31) == 0) { red[w] = s; red[8 + w] = s2; }
  __syncthreads();
  s = 0.f; s2 = 0.f;
  #pragma unroll
  for (int i = 0; i < 8; ++i) { s += red[i]; s2 += red[8 + i]; }
  const float mean = s * (1.0f / E_DIM);
  const float var  = s2 * (1.0f / E_DIM) - mean * mean;
  const float rstd = rsqrtf(var + 1e-5f);
  float* yr = y + (size_t)row * E_DIM;
  for (int i = threadIdx.x; i < E_DIM; i += 256)
    yr[i] = (xr[i] - mean) * rstd * g[i] + b[i];
}

// ---------------------------------------------------------------------------
// Tiled WMMA GEMM: C = act(A(MxK) @ B(KxN) + bias + resid)
// 128x128 block tile, BK=32, 8 waves of 32x64.
// Double-buffered LDS: global loads for tile t+1 are issued before the 8-WMMA
// burst on tile t, so HBM/L2 latency hides under the matrix pipe; the staged
// registers are converted to bf16 and ds_stored afterwards. One barrier per
// K-step. Requires K % 32 == 0 (true for all uses: K in {1024, 4096}).
// ---------------------------------------------------------------------------
#define GA_LD 40   // 32 + 8 pad (bf16 elems)
#define GB_LD 136  // 128 + 8 pad

template <bool HAS_BIAS, bool HAS_RES, bool RELU>
__global__ __launch_bounds__(256)
void gemm_kernel(const float* __restrict__ A, const float* __restrict__ B,
                 const float* __restrict__ bias, const float* __restrict__ resid,
                 float* __restrict__ C, int M, int N, int K) {
  __shared__ __align__(16) bf16 Alds[2][128 * GA_LD];
  __shared__ __align__(16) bf16 Blds[2][32 * GB_LD];

  const int tid  = threadIdx.x;
  const int lane = tid & 31;
  const int wave = tid >> 5;
  const int wm   = (wave >> 1) * 32;   // wave M offset in tile (0,32,64,96)
  const int wn   = (wave & 1) * 64;    // wave N offset in tile (0,64)
  const int mBase = blockIdx.y * 128;
  const int nBase = blockIdx.x * 128;
  const int half = lane >> 4;
  const int nl   = lane & 15;

  // staging coordinates
  const int ar = tid >> 3;             // A row 0..31 (+32*p)
  const int ac = (tid & 7) * 4;        // A col 0..28
  const int br = tid >> 5;             // B row 0..7  (+8*p)
  const int bc = (tid & 31) * 4;       // B col 0..124

  float4 fa[4];
  float  fb[4][4];

  auto loadTile = [&](int k0, bool pf) {
    #pragma unroll
    for (int p = 0; p < 4; ++p) {
      const int gm = mBase + ar + p * 32;
      fa[p] = make_float4(0.f, 0.f, 0.f, 0.f);
      if (gm < M) {
        const float* src = A + (size_t)gm * K + k0 + ac;
        fa[p] = *(const float4*)src;
        if (pf) __builtin_prefetch(src + 32, 0, 3);   // warm L2 for tile t+2
      }
    }
    #pragma unroll
    for (int p = 0; p < 4; ++p) {
      const size_t brow = (size_t)(k0 + br + p * 8) * N;
      #pragma unroll
      for (int u = 0; u < 4; ++u) {
        const int gn = nBase + bc + u;
        // scalar loads: head GEMM rows are not 16B aligned (V=50257 odd)
        fb[p][u] = (gn < N) ? B[brow + gn] : 0.0f;
      }
      if (pf) {
        const int gn = nBase + bc;
        if (gn < N) __builtin_prefetch(B + brow + 32u * N + gn, 0, 3);
      }
    }
  };

  auto storeStage = [&](int buf) {
    #pragma unroll
    for (int p = 0; p < 4; ++p) {
      bf16* d = &Alds[buf][(ar + p * 32) * GA_LD + ac];
      d[0] = (bf16)fa[p].x; d[1] = (bf16)fa[p].y;
      d[2] = (bf16)fa[p].z; d[3] = (bf16)fa[p].w;
      bf16* e = &Blds[buf][(br + p * 8) * GB_LD + bc];
      e[0] = (bf16)fb[p][0]; e[1] = (bf16)fb[p][1];
      e[2] = (bf16)fb[p][2]; e[3] = (bf16)fb[p][3];
    }
  };

  const v8f vzero = {0.f, 0.f, 0.f, 0.f, 0.f, 0.f, 0.f, 0.f};
  v8f acc[2][4];
  #pragma unroll
  for (int i = 0; i < 2; ++i)
    #pragma unroll
    for (int j = 0; j < 4; ++j) acc[i][j] = vzero;

  // prologue: stage tile 0 into buffer 0
  loadTile(0, false);
  storeStage(0);

  int cur = 0;
  for (int k0 = 0; k0 < K; k0 += 32) {
    __syncthreads();                       // buf[cur] staged; buf[cur^1] free
    const bool more = (k0 + 32) < K;
    if (more) loadTile(k0 + 32, (k0 + 64) < K);   // long-latency loads first

    // hoist all fragment loads ahead of the WMMA burst
    const v16bf a0 = load_frag_a(&Alds[cur][(wm +  0) * GA_LD], GA_LD);
    const v16bf a1 = load_frag_a(&Alds[cur][(wm + 16) * GA_LD], GA_LD);
    const v16bf b0 = load_frag_b(Blds[cur], GB_LD, wn +  0);
    const v16bf b1 = load_frag_b(Blds[cur], GB_LD, wn + 16);
    const v16bf b2 = load_frag_b(Blds[cur], GB_LD, wn + 32);
    const v16bf b3 = load_frag_b(Blds[cur], GB_LD, wn + 48);

    acc[0][0] = wmma_bf16(a0, b0, acc[0][0]);
    acc[1][0] = wmma_bf16(a1, b0, acc[1][0]);
    acc[0][1] = wmma_bf16(a0, b1, acc[0][1]);
    acc[1][1] = wmma_bf16(a1, b1, acc[1][1]);
    acc[0][2] = wmma_bf16(a0, b2, acc[0][2]);
    acc[1][2] = wmma_bf16(a1, b2, acc[1][2]);
    acc[0][3] = wmma_bf16(a0, b3, acc[0][3]);
    acc[1][3] = wmma_bf16(a1, b3, acc[1][3]);

    if (more) storeStage(cur ^ 1);         // convert + ds_store next tile
    cur ^= 1;
  }

  // --- fused epilogue: bias + residual + relu (compile-time selected) ---
  #pragma unroll
  for (int i = 0; i < 2; ++i) {
    #pragma unroll
    for (int j = 0; j < 4; ++j) {
      #pragma unroll
      for (int r = 0; r < 8; ++r) {
        const int m = mBase + wm + 16 * i + r + 8 * half;
        const int n = nBase + wn + 16 * j + nl;
        if (m < M && n < N) {
          float v = acc[i][j][r];
          if (HAS_BIAS) v += bias[n];
          if (HAS_RES)  v += resid[(size_t)m * N + n];
          if (RELU)     v = fmaxf(v, 0.0f);
          C[(size_t)m * N + n] = v;
        }
      }
    }
  }
}

// ---------------------------------------------------------------------------
// Attention: one block per (b,h). Stages q/k^T/v as bf16 in LDS, WMMA scores,
// in-register causal softmax (cross-lane shfl within 16-lane halves), WMMA
// att@V. att probabilities reuse the q/k^T LDS region after a barrier.
// ---------------------------------------------------------------------------
#define QS_LD  72   // 64 + 8
#define KT_LD 136   // 128 + 8
#define VS_LD  72
#define AT_LD 136
// smem layout (bf16 elems): q[0,9216) kt[9216,17920) v[17920,27136); att reuses [0,17408)
#define SM_TOTAL 27136

__global__ __launch_bounds__(256)
void attn_kernel(const float* __restrict__ Q, const float* __restrict__ K,
                 const float* __restrict__ V, float* __restrict__ O) {
  __shared__ __align__(16) bf16 smem[SM_TOTAL];
  bf16* q_l   = smem;
  bf16* kt_l  = smem + 9216;
  bf16* v_l   = smem + 17920;
  bf16* att_l = smem;                 // overlaps q/kt after barrier

  const int bh = blockIdx.x;
  const int b  = bh >> 4;
  const int h  = bh & 15;
  const size_t base = (size_t)b * T_SEQ * E_DIM + (size_t)h * D_HEAD;

  const int tid = threadIdx.x;
  #pragma unroll
  for (int p = 0; p < 8; ++p) {
    const int r = (tid >> 4) + p * 16;       // 0..127
    const int c = (tid & 15) * 4;            // 0..60
    const size_t g = base + (size_t)r * E_DIM + c;
    const float4 fq = *(const float4*)(Q + g);
    const float4 fk = *(const float4*)(K + g);
    const float4 fv = *(const float4*)(V + g);
    bf16* qd = &q_l[r * QS_LD + c];
    qd[0] = (bf16)fq.x; qd[1] = (bf16)fq.y; qd[2] = (bf16)fq.z; qd[3] = (bf16)fq.w;
    bf16* vd = &v_l[r * VS_LD + c];
    vd[0] = (bf16)fv.x; vd[1] = (bf16)fv.y; vd[2] = (bf16)fv.z; vd[3] = (bf16)fv.w;
    kt_l[(c + 0) * KT_LD + r] = (bf16)fk.x;
    kt_l[(c + 1) * KT_LD + r] = (bf16)fk.y;
    kt_l[(c + 2) * KT_LD + r] = (bf16)fk.z;
    kt_l[(c + 3) * KT_LD + r] = (bf16)fk.w;
  }
  __syncthreads();

  const int lane  = tid & 31;
  const int wave  = tid >> 5;
  const int tBase = wave * 16;               // 8 waves cover 128 rows
  const int half  = lane >> 4;
  const int nl    = lane & 15;

  const v8f vzero = {0.f, 0.f, 0.f, 0.f, 0.f, 0.f, 0.f, 0.f};
  v8f sacc[8];
  #pragma unroll
  for (int nf = 0; nf < 8; ++nf) sacc[nf] = vzero;

  // scores = q @ k^T  (K loop over d=64 in 2 WMMA steps)
  #pragma unroll
  for (int d0 = 0; d0 < D_HEAD; d0 += 32) {
    const v16bf a = load_frag_a(&q_l[tBase * QS_LD + d0], QS_LD);
    #pragma unroll
    for (int nf = 0; nf < 8; ++nf) {
      const v16bf bf = load_frag_b(&kt_l[d0 * KT_LD], KT_LD, nf * 16);
      sacc[nf] = wmma_bf16(a, bf, sacc[nf]);
    }
  }
  __syncthreads();   // all q/kt reads done; att_l may now overwrite them

  // causal softmax, scale = E^-0.5 = 1/32 (reference uses full n_embd)
  const float scale = 0.03125f;
  #pragma unroll
  for (int r = 0; r < 8; ++r) {
    const int t = tBase + r + 8 * half;
    float e[8];
    float mx = -3.0e38f;
    #pragma unroll
    for (int nf = 0; nf < 8; ++nf) {
      const int s = nf * 16 + nl;
      float vv = sacc[nf][r] * scale;
      vv = (s <= t) ? vv : -3.0e38f;
      e[nf] = vv;
      mx = fmaxf(mx, vv);
    }
    #pragma unroll
    for (int m = 1; m < 16; m <<= 1) mx = fmaxf(mx, __shfl_xor(mx, m, 32));
    float sum = 0.f;
    #pragma unroll
    for (int nf = 0; nf < 8; ++nf) { float ex = __expf(e[nf] - mx); e[nf] = ex; sum += ex; }
    #pragma unroll
    for (int m = 1; m < 16; m <<= 1) sum += __shfl_xor(sum, m, 32);
    const float inv = 1.0f / sum;
    #pragma unroll
    for (int nf = 0; nf < 8; ++nf)
      att_l[(size_t)t * AT_LD + nf * 16 + nl] = (bf16)(e[nf] * inv);
  }
  __syncthreads();

  // out = att @ v  (K loop over s=128 in 4 WMMA steps)
  v8f oacc[4];
  #pragma unroll
  for (int j = 0; j < 4; ++j) oacc[j] = vzero;
  #pragma unroll
  for (int s0 = 0; s0 < T_SEQ; s0 += 32) {
    const v16bf a = load_frag_a(&att_l[tBase * AT_LD + s0], AT_LD);
    #pragma unroll
    for (int j = 0; j < 4; ++j) {
      const v16bf bf = load_frag_b(&v_l[s0 * VS_LD], VS_LD, j * 16);
      oacc[j] = wmma_bf16(a, bf, oacc[j]);
    }
  }
  #pragma unroll
  for (int j = 0; j < 4; ++j) {
    #pragma unroll
    for (int r = 0; r < 8; ++r) {
      const int t = tBase + r + 8 * half;
      O[base + (size_t)t * E_DIM + j * 16 + nl] = oacc[j][r];
    }
  }
}

// ---------------------------------------------------------------------------
// Per-row NLL: nll[row] = logsumexp(logits[row]) - logits[row][target]
// ---------------------------------------------------------------------------
__global__ __launch_bounds__(256)
void nll_kernel(const float* __restrict__ logits, const int* __restrict__ tgt,
                float* __restrict__ nll) {
  __shared__ float red[8];
  const int row = blockIdx.x;
  const float* lr = logits + (size_t)row * V_DIM;
  float mx = -3.4e38f;
  for (int i = threadIdx.x; i < V_DIM; i += 256) mx = fmaxf(mx, lr[i]);
  #pragma unroll
  for (int m = 16; m >= 1; m >>= 1) mx = fmaxf(mx, __shfl_xor(mx, m, 32));
  if ((threadIdx.x & 31) == 0) red[threadIdx.x >> 5] = mx;
  __syncthreads();
  mx = red[0];
  #pragma unroll
  for (int i = 1; i < 8; ++i) mx = fmaxf(mx, red[i]);
  __syncthreads();
  float s = 0.f;
  for (int i = threadIdx.x; i < V_DIM; i += 256) s += __expf(lr[i] - mx);
  #pragma unroll
  for (int m = 16; m >= 1; m >>= 1) s += __shfl_xor(s, m, 32);
  if ((threadIdx.x & 31) == 0) red[threadIdx.x >> 5] = s;
  __syncthreads();
  if (threadIdx.x == 0) {
    float tot = 0.f;
    #pragma unroll
    for (int i = 0; i < 8; ++i) tot += red[i];
    nll[row] = (logf(tot) + mx) - lr[tgt[row]];
  }
}

__global__ __launch_bounds__(256)
void mean_kernel(const float* __restrict__ nll, float* __restrict__ out) {
  __shared__ float red[8];
  float s = 0.f;
  for (int i = threadIdx.x; i < M_ROWS; i += 256) s += nll[i];
  #pragma unroll
  for (int m = 16; m >= 1; m >>= 1) s += __shfl_xor(s, m, 32);
  if ((threadIdx.x & 31) == 0) red[threadIdx.x >> 5] = s;
  __syncthreads();
  if (threadIdx.x == 0) {
    float tot = 0.f;
    #pragma unroll
    for (int i = 0; i < 8; ++i) tot += red[i];
    out[0] = tot * (1.0f / M_ROWS);
  }
}

// ---------------------------------------------------------------------------
// Orchestration
// ---------------------------------------------------------------------------
static inline void launch_gemm(const float* A, const float* B, const float* bias,
                               const float* resid, float* C,
                               int M, int N, int K, int relu, hipStream_t s) {
  dim3 grid((N + 127) / 128, (M + 127) / 128);
  if (relu) {
    gemm_kernel<true, false, true><<<grid, 256, 0, s>>>(A, B, bias, resid, C, M, N, K);
  } else if (bias && resid) {
    gemm_kernel<true, true, false><<<grid, 256, 0, s>>>(A, B, bias, resid, C, M, N, K);
  } else if (bias) {
    gemm_kernel<true, false, false><<<grid, 256, 0, s>>>(A, B, bias, resid, C, M, N, K);
  } else {
    gemm_kernel<false, false, false><<<grid, 256, 0, s>>>(A, B, bias, resid, C, M, N, K);
  }
}

extern "C" void kernel_launch(void* const* d_in, const int* in_sizes, int n_in,
                              void* d_out, int out_size, void* d_ws, size_t ws_size,
                              hipStream_t stream) {
  (void)in_sizes; (void)n_in; (void)out_size; (void)ws_size;

  const int*   idx     = (const int*)d_in[0];
  const int*   targets = (const int*)d_in[1];
  const float* tok_emb = (const float*)d_in[2];
  const float* pos_emb = (const float*)d_in[3];
  const float* wq      = (const float*)d_in[4];
  const float* wk      = (const float*)d_in[5];
  const float* wv      = (const float*)d_in[6];
  const float* w_proj  = (const float*)d_in[7];
  const float* b_proj  = (const float*)d_in[8];
  const float* ln1_g   = (const float*)d_in[9];
  const float* ln1_b   = (const float*)d_in[10];
  const float* ln2_g   = (const float*)d_in[11];
  const float* ln2_b   = (const float*)d_in[12];
  const float* w_fc1   = (const float*)d_in[13];
  const float* b_fc1   = (const float*)d_in[14];
  const float* w_fc2   = (const float*)d_in[15];
  const float* b_fc2   = (const float*)d_in[16];
  const float* lnf_g   = (const float*)d_in[17];
  const float* lnf_b   = (const float*)d_in[18];
  const float* w_head  = (const float*)d_in[19];
  const float* b_head  = (const float*)d_in[20];

  // Workspace layout (floats). Total ~84 MB.
  float* ws = (float*)d_ws;
  const size_t ME = (size_t)M_ROWS * E_DIM;       // 2M floats
  float* x   = ws;                                // residual stream
  float* h   = ws + 1 * ME;                       // LN output
  float* q   = ws + 2 * ME;
  float* k   = ws + 3 * ME;
  float* v   = ws + 4 * ME;
  float* o   = ws + 5 * ME;                       // attention output
  float* ff  = ws + 6 * ME;                       // 2048x4096 (uses 4*ME)
  float* nll = ws + 10 * ME;                      // 2048 floats

  float* logits = (float*)d_out;                  // (B*T, V) fp32
  float* loss   = logits + (size_t)M_ROWS * V_DIM;

  // x = tok_emb[idx] + pos_emb
  embed_kernel<<<M_ROWS, 256, 0, stream>>>(idx, tok_emb, pos_emb, x);

  for (int l = 0; l < L_NUM; ++l) {
    const size_t oEE = (size_t)l * E_DIM * E_DIM;
    const size_t oEF = (size_t)l * E_DIM * FF_DIM;
    const size_t oE  = (size_t)l * E_DIM;
    const size_t oF  = (size_t)l * FF_DIM;

    // h = LN1(x)
    layernorm_kernel<<<M_ROWS, 256, 0, stream>>>(x, ln1_g + oE, ln1_b + oE, h);
    // q,k,v = h @ W{q,k,v}
    launch_gemm(h, wq + oEE, nullptr, nullptr, q, M_ROWS, E_DIM, E_DIM, 0, stream);
    launch_gemm(h, wk + oEE, nullptr, nullptr, k, M_ROWS, E_DIM, E_DIM, 0, stream);
    launch_gemm(h, wv + oEE, nullptr, nullptr, v, M_ROWS, E_DIM, E_DIM, 0, stream);
    // o = softmax(causal(q k^T / 32)) v   (per (b,h))
    attn_kernel<<<B_SZ * H_NUM, 256, 0, stream>>>(q, k, v, o);
    // x = x + o @ w_proj + b_proj   (fused residual epilogue; in-place safe)
    launch_gemm(o, w_proj + oEE, b_proj + oE, x, x, M_ROWS, E_DIM, E_DIM, 0, stream);
    // h = LN2(x)
    layernorm_kernel<<<M_ROWS, 256, 0, stream>>>(x, ln2_g + oE, ln2_b + oE, h);
    // ff = relu(h @ w_fc1 + b_fc1)
    launch_gemm(h, w_fc1 + oEF, b_fc1 + oF, nullptr, ff, M_ROWS, FF_DIM, E_DIM, 1, stream);
    // x = x + ff @ w_fc2 + b_fc2
    launch_gemm(ff, w_fc2 + oEF, b_fc2 + oE, x, x, M_ROWS, E_DIM, FF_DIM, 0, stream);
  }

  // final LN + vocab head (writes logits straight into d_out)
  layernorm_kernel<<<M_ROWS, 256, 0, stream>>>(x, lnf_g, lnf_b, h);
  launch_gemm(h, w_head, b_head, nullptr, logits, M_ROWS, V_DIM, E_DIM, 0, stream);

  // loss = mean over rows of (logsumexp - logit[target])
  nll_kernel<<<M_ROWS, 256, 0, stream>>>(logits, targets, nll);
  mean_kernel<<<1, 256, 0, stream>>>(nll, loss);
}
